// Swin3DLayer_82497731821878
// MI455X (gfx1250) — matrix operand order repaired and checked by reading, not compiled, not run on hardware
//
#include <hip/hip_runtime.h>
#include <hip/hip_bf16.h>
#include <cmath>

typedef _Float16 v16h __attribute__((ext_vector_type(16)));
typedef _Float16 h8   __attribute__((ext_vector_type(8)));
typedef _Float16 h4   __attribute__((ext_vector_type(4)));
typedef float    v8f  __attribute__((ext_vector_type(8)));

#define NTOK   100352        // B*D*H*W
#define NSPAT  50176         // D*H*W
#define NWIN   1568          // B * 784
#define NWB    784           // windows per batch (4*14*14)

// ---------------------------------------------------------------------------
// WMMA helpers (gfx1250, wave32)
// ---------------------------------------------------------------------------
__device__ __forceinline__ v8f wmma32(v16h a, v16h b, v8f c) {
  return __builtin_amdgcn_wmma_f32_16x16x32_f16(false, a, false, b,
                                                (short)0, c, false, false);
}

// A fragment: 16x32 f16.  lanes0-15: M=lane, elems{0..7}=K0..7, {8..15}=K16..23
//             lanes16-31: M=lane-16,  elems{0..7}=K8..15,{8..15}=K24..31
__device__ __forceinline__ v16h load_afrag(const _Float16* base, int ld, int lane) {
  int m  = lane & 15;
  int kb = (lane < 16) ? 0 : 8;
  const _Float16* p = base + m * ld + kb;
  h8 lo = *(const h8*)(p);
  h8 hi = *(const h8*)(p + 16);
  v16h f;
#pragma unroll
  for (int i = 0; i < 8; ++i) { f[i] = lo[i]; f[i + 8] = hi[i]; }
  return f;
}

// B fragment: 32x16 f16, stored n-major in LDS (row n holds K 0..31 contiguous)
// lanes0-15: N=lane, K=0..7,16..23 ; lanes16-31: N=lane-16, K=8..15,24..31
__device__ __forceinline__ v16h load_bfrag(const _Float16* base, int ld, int lane) {
  int n  = lane & 15;
  int kb = (lane < 16) ? 0 : 8;
  const _Float16* p = base + n * ld + kb;
  h8 lo = *(const h8*)(p);
  h8 hi = *(const h8*)(p + 16);
  v16h f;
#pragma unroll
  for (int i = 0; i < 8; ++i) { f[i] = lo[i]; f[i + 8] = hi[i]; }
  return f;
}

__device__ __forceinline__ float gelu_exact(float x) {
  return 0.5f * x * (1.0f + erff(x * 0.70710678118654752f));
}

// ---------------------------------------------------------------------------
// Layout transposes: (B,C,S) <-> (B,S,C), tiled through LDS
// ---------------------------------------------------------------------------
__global__ __launch_bounds__(256) void to_tokens(const float* __restrict__ in,
                                                 float* __restrict__ out) {
  __shared__ float tile[32][33];
  int b  = blockIdx.z;
  int s0 = blockIdx.x * 32, c0 = blockIdx.y * 32;
  int tx = threadIdx.x, ty = threadIdx.y;           // 32 x 8
  const float* ib = in + (size_t)b * 128 * NSPAT;
#pragma unroll
  for (int i = 0; i < 32; i += 8)
    tile[ty + i][tx] = ib[(size_t)(c0 + ty + i) * NSPAT + s0 + tx];
  __syncthreads();
  float* ob = out + (size_t)b * NSPAT * 128;
#pragma unroll
  for (int i = 0; i < 32; i += 8)
    ob[(size_t)(s0 + ty + i) * 128 + c0 + tx] = tile[tx][ty + i];
}

__global__ __launch_bounds__(256) void from_tokens(const float* __restrict__ in,
                                                   float* __restrict__ out) {
  __shared__ float tile[32][33];
  int b  = blockIdx.z;
  int s0 = blockIdx.x * 32, c0 = blockIdx.y * 32;
  int tx = threadIdx.x, ty = threadIdx.y;
  const float* ib = in + (size_t)b * NSPAT * 128;
#pragma unroll
  for (int i = 0; i < 32; i += 8)
    tile[ty + i][tx] = ib[(size_t)(s0 + ty + i) * 128 + c0 + tx];
  __syncthreads();
  float* ob = out + (size_t)b * 128 * NSPAT;
#pragma unroll
  for (int i = 0; i < 32; i += 8)
    ob[(size_t)(c0 + ty + i) * NSPAT + s0 + tx] = tile[tx][ty + i];
}

// f32 -> f16 weight cast (weights (N,K) row-major are already WMMA-B n-major)
__global__ __launch_bounds__(256) void cvt_f16(const float* __restrict__ in,
                                               _Float16* __restrict__ out, int n) {
  int i = blockIdx.x * 256 + threadIdx.x;
  if (i < n) out[i] = (_Float16)in[i];
}

// ---------------------------------------------------------------------------
// LayerNorm over C=128: one wave per token, 4 channels per lane
// ---------------------------------------------------------------------------
__global__ __launch_bounds__(256) void ln_kernel(const float* __restrict__ x,
                                                 const float* __restrict__ g,
                                                 const float* __restrict__ b,
                                                 _Float16* __restrict__ out) {
  int token = blockIdx.x * 8 + (threadIdx.x >> 5);
  int lane  = threadIdx.x & 31;
  float4 v = *(const float4*)(x + (size_t)token * 128 + lane * 4);
  float s = v.x + v.y + v.z + v.w;
#pragma unroll
  for (int o = 16; o; o >>= 1) s += __shfl_xor(s, o, 32);
  float mu = s * (1.0f / 128.0f);
  float d0 = v.x - mu, d1 = v.y - mu, d2 = v.z - mu, d3 = v.w - mu;
  float vs = d0 * d0 + d1 * d1 + d2 * d2 + d3 * d3;
#pragma unroll
  for (int o = 16; o; o >>= 1) vs += __shfl_xor(vs, o, 32);
  float rstd = rsqrtf(vs * (1.0f / 128.0f) + 1e-5f);
  int c = lane * 4;
  h4 r;
  r[0] = (_Float16)(d0 * rstd * g[c + 0] + b[c + 0]);
  r[1] = (_Float16)(d1 * rstd * g[c + 1] + b[c + 1]);
  r[2] = (_Float16)(d2 * rstd * g[c + 2] + b[c + 2]);
  r[3] = (_Float16)(d3 * rstd * g[c + 3] + b[c + 3]);
  *(h4*)(out + (size_t)token * 128 + c) = r;
}

// ---------------------------------------------------------------------------
// Generic WMMA GEMM:  Y[m,n] = act( sum_k A[m,k]*W[n,k] + bias[n] ) (+res)
// BM=256 BN=64 BK=32, 256 threads = 8 waves; wave computes 32x64
// (two A-fragments amortize each B-fragment -> 8 WMMA per K step per wave)
// ---------------------------------------------------------------------------
template <int N, int K, int ACT, bool RES, bool BIAS, bool OUTF16>
__global__ __launch_bounds__(256) void gemm_ws(const _Float16* __restrict__ A,
                                               const _Float16* __restrict__ W,
                                               const float* __restrict__ bias,
                                               const float* __restrict__ res,
                                               _Float16* __restrict__ o16,
                                               float* __restrict__ o32) {
  __shared__ _Float16 As[256 * 40];   // padded rows: conflict-free b128 frag reads
  __shared__ _Float16 Bs[64 * 40];
  const int tid  = threadIdx.x;
  const int lane = tid & 31, wave = tid >> 5;
  const int m0 = blockIdx.x * 256;
  const int n0 = blockIdx.y * 64;
  v8f acc[2][4] = {{v8f{}, v8f{}, v8f{}, v8f{}}, {v8f{}, v8f{}, v8f{}, v8f{}}};

#pragma unroll
  for (int kb = 0; kb < K; kb += 32) {
    {   // A tile 256x32: one row (32 halves) per thread
      const h8* src = (const h8*)(A + (size_t)(m0 + tid) * K + kb);
      h8 a0 = src[0], a1 = src[1], a2 = src[2], a3 = src[3];
      _Float16* d = As + tid * 40;
      *(h8*)(d)      = a0;
      *(h8*)(d + 8)  = a1;
      *(h8*)(d + 16) = a2;
      *(h8*)(d + 24) = a3;
    }
    {   // W tile 64(N) x 32(K): 8 halves per thread
      int r = tid >> 2, c = (tid & 3) * 8;
      h8 w = *(const h8*)(W + (size_t)(n0 + r) * K + kb + c);
      *(h8*)(Bs + r * 40 + c) = w;
    }
    __syncthreads();
    v16h af0 = load_afrag(As + (wave * 32) * 40,      40, lane);
    v16h af1 = load_afrag(As + (wave * 32 + 16) * 40, 40, lane);
#pragma unroll
    for (int nt = 0; nt < 4; ++nt) {
      v16h bf = load_bfrag(Bs + nt * 16 * 40, 40, lane);
      acc[0][nt] = wmma32(af0, bf, acc[0][nt]);
      acc[1][nt] = wmma32(af1, bf, acc[1][nt]);
    }
    __syncthreads();
  }

  int nn  = lane & 15;
  int off = (lane >> 4) * 8;
#pragma unroll
  for (int mi = 0; mi < 2; ++mi) {
#pragma unroll
    for (int nt = 0; nt < 4; ++nt) {
#pragma unroll
      for (int r = 0; r < 8; ++r) {
        int m = m0 + wave * 32 + mi * 16 + r + off;
        int n = n0 + nt * 16 + nn;
        float v = acc[mi][nt][r];
        if (BIAS) v += bias[n];
        if (ACT == 1) v = gelu_exact(v);
        if (RES) v += res[(size_t)m * N + n];
        if (OUTF16) o16[(size_t)m * N + n] = (_Float16)v;
        else        o32[(size_t)m * N + n] = v;
      }
    }
  }
}

// ---------------------------------------------------------------------------
// Windowed attention: one wave per (window, head), SHIFTED is compile-time.
// Shift/partition folded into the gather map; rel-pos bias staged in LDS and
// applied (with the shift mask) during the f32 softmax pass, so the WMMA
// score epilogue is just mul + ds_store.
// ---------------------------------------------------------------------------
template <int SHIFTED>
__global__ __launch_bounds__(32) void attn_kernel(const _Float16* __restrict__ qkv,
                                                  const float* __restrict__ rpb,
                                                  _Float16* __restrict__ out) {
  __shared__ _Float16 ql[64 * 40];
  __shared__ _Float16 kl[64 * 40];
  __shared__ _Float16 vt[32 * 72];   // v transposed: row = channel, col = key
  __shared__ float    Sm[64 * 68];   // scores
  __shared__ _Float16 Pl[64 * 72];   // softmax probs (f16)
  __shared__ float    bl[344];       // per-head rel-pos bias table
  __shared__ int      gidx[64];
  __shared__ int      rlab[64];

  const int lane = threadIdx.x;
  const int h    = blockIdx.y;
  const int win  = blockIdx.x;
  const int b  = win / NWB;
  const int r  = win % NWB;
  const int wd = r / 196, wh = (r / 14) % 14, ww = r % 14;

  // stage this head's bias table
  for (int i = lane; i < 343; i += 32) bl[i] = rpb[i * 4 + h];

  // token -> global index (+ region label for the shift mask)
#pragma unroll
  for (int t = lane; t < 64; t += 32) {
    int td = t >> 4, th = (t >> 2) & 3, tw = t & 3;
    int dp = wd * 4 + td, hp = wh * 4 + th, wp = ww * 4 + tw;  // shifted-space
    int d_ = dp, h_ = hp, w_ = wp;
    if (SHIFTED) { d_ = (dp + 2) & 15; h_ = (hp + 2) % 56; w_ = (wp + 2) % 56; }
    gidx[t] = ((b * 16 + d_) * 56 + h_) * 56 + w_;
    int ad = dp < 12 ? 0 : (dp < 14 ? 1 : 2);
    int ah = hp < 52 ? 0 : (hp < 54 ? 1 : 2);
    int aw = wp < 52 ? 0 : (wp < 54 ? 1 : 2);
    rlab[t] = (ad * 3 + ah) * 3 + aw;
  }
  __syncthreads();

  // stage q, k, v^T for this head
#pragma unroll
  for (int t = lane; t < 64; t += 32) {
    const _Float16* row = qkv + (size_t)gidx[t] * 384 + h * 32;
    *(h8*)(ql + t * 40)      = *(const h8*)(row);
    *(h8*)(ql + t * 40 + 8)  = *(const h8*)(row + 8);
    *(h8*)(ql + t * 40 + 16) = *(const h8*)(row + 16);
    *(h8*)(ql + t * 40 + 24) = *(const h8*)(row + 24);
    const _Float16* krow = row + 128;
    *(h8*)(kl + t * 40)      = *(const h8*)(krow);
    *(h8*)(kl + t * 40 + 8)  = *(const h8*)(krow + 8);
    *(h8*)(kl + t * 40 + 16) = *(const h8*)(krow + 16);
    *(h8*)(kl + t * 40 + 24) = *(const h8*)(krow + 24);
    const _Float16* vrow = row + 256;
    h8 vv[4];
    vv[0] = *(const h8*)(vrow);      vv[1] = *(const h8*)(vrow + 8);
    vv[2] = *(const h8*)(vrow + 16); vv[3] = *(const h8*)(vrow + 24);
#pragma unroll
    for (int cc = 0; cc < 4; ++cc)
#pragma unroll
      for (int i = 0; i < 8; ++i)
        vt[(cc * 8 + i) * 72 + t] = vv[cc][i];
  }
  __syncthreads();

  // S = scale * (q k^T); hd = 32 -> single K step. B-fragments hoisted.
  const float scale = 0.17677669529663687f;   // 32^-0.5
  const int nn  = lane & 15;
  const int off = (lane >> 4) * 8;
  {
    v16h bf0 = load_bfrag(kl + 0 * 16 * 40, 40, lane);
    v16h bf1 = load_bfrag(kl + 1 * 16 * 40, 40, lane);
    v16h bf2 = load_bfrag(kl + 2 * 16 * 40, 40, lane);
    v16h bf3 = load_bfrag(kl + 3 * 16 * 40, 40, lane);
#pragma unroll
    for (int mt = 0; mt < 4; ++mt) {
      v16h af = load_afrag(ql + mt * 16 * 40, 40, lane);
      v8f z = v8f{};
      v8f a0 = wmma32(af, bf0, z);
      v8f a1 = wmma32(af, bf1, z);
      v8f a2 = wmma32(af, bf2, z);
      v8f a3 = wmma32(af, bf3, z);
#pragma unroll
      for (int r2 = 0; r2 < 8; ++r2) {
        int m = mt * 16 + r2 + off;
        float* srow = Sm + m * 68 + nn;
        srow[0]  = a0[r2] * scale;
        srow[16] = a1[r2] * scale;
        srow[32] = a2[r2] * scale;
        srow[48] = a3[r2] * scale;
      }
    }
  }
  __syncthreads();

  // softmax with fused bias (+mask): each lane owns 2 full rows
#pragma unroll
  for (int rr = 0; rr < 2; ++rr) {
    int m = lane + rr * 32;
    float* srow = Sm + m * 68;
    // bias index: rp = mbase - (tdj*49 + thj*7 + twj)
    int mbase = ((m >> 4) + 3) * 49 + (((m >> 2) & 3) + 3) * 7 + ((m & 3) + 3);
    int labm  = rlab[m];
    float mx = -3.0e38f;
#pragma unroll 4
    for (int j = 0; j < 64; ++j) {
      int jo = (j >> 4) * 49 + ((j >> 2) & 3) * 7 + (j & 3);
      float v = srow[j] + bl[mbase - jo];
      if (SHIFTED) v = (rlab[j] == labm) ? v : (v - 100.0f);
      srow[j] = v;
      mx = fmaxf(mx, v);
    }
    float sum = 0.0f;
#pragma unroll 4
    for (int j = 0; j < 64; ++j) {
      float e = __expf(srow[j] - mx);
      srow[j] = e;
      sum += e;
    }
    float inv = 1.0f / sum;
    _Float16* prow = Pl + m * 72;
#pragma unroll 4
    for (int j = 0; j < 64; ++j) prow[j] = (_Float16)(srow[j] * inv);
  }
  __syncthreads();

  // O = P * V  (64x64 x 64x32): K = 64 -> two K steps. V fragments hoisted.
  {
    v16h vb[2][2];
#pragma unroll
    for (int nt = 0; nt < 2; ++nt)
#pragma unroll
      for (int ks = 0; ks < 2; ++ks)
        vb[nt][ks] = load_bfrag(vt + nt * 16 * 72 + ks * 32, 72, lane);
#pragma unroll
    for (int mt = 0; mt < 4; ++mt) {
      v16h af0 = load_afrag(Pl + mt * 16 * 72,      72, lane);
      v16h af1 = load_afrag(Pl + mt * 16 * 72 + 32, 72, lane);
#pragma unroll
      for (int nt = 0; nt < 2; ++nt) {
        v8f acc = v8f{};
        acc = wmma32(af0, vb[nt][0], acc);
        acc = wmma32(af1, vb[nt][1], acc);
#pragma unroll
        for (int r2 = 0; r2 < 8; ++r2) {
          int m = mt * 16 + r2 + off;
          int n = nt * 16 + nn;
          out[(size_t)gidx[m] * 128 + h * 32 + n] = (_Float16)acc[r2];
        }
      }
    }
  }
}

// ---------------------------------------------------------------------------
// Host launcher
// ---------------------------------------------------------------------------
extern "C" void kernel_launch(void* const* d_in, const int* in_sizes, int n_in,
                              void* d_out, int out_size, void* d_ws, size_t ws_size,
                              hipStream_t stream) {
  (void)in_sizes; (void)n_in; (void)out_size; (void)ws_size;
  const float* x = (const float*)d_in[0];

  // workspace carve-out
  size_t off = 0;
  auto alloc = [&](size_t bytes) {
    void* p = (char*)d_ws + off;
    off += (bytes + 255) & ~(size_t)255;
    return p;
  };
  const size_t T = NTOK;
  float*    xt  = (float*)   alloc(T * 128 * 4);   // current x (token-major)
  float*    xr  = (float*)   alloc(T * 128 * 4);   // post-attention residual
  _Float16* xn  = (_Float16*)alloc(T * 128 * 2);   // LN output (f16)
  _Float16* ao  = (_Float16*)alloc(T * 128 * 2);   // attention output
  _Float16* big = (_Float16*)alloc(T * 512 * 2);   // qkv (T*384) / fc1 hidden (T*512)
  _Float16* wq  = (_Float16*)alloc(384 * 128 * 2);
  _Float16* wp  = (_Float16*)alloc(128 * 128 * 2);
  _Float16* w1  = (_Float16*)alloc(512 * 128 * 2);
  _Float16* w2  = (_Float16*)alloc(128 * 512 * 2);

  // (B,C,D,H,W) -> (T,128)
  to_tokens<<<dim3(NSPAT / 32, 4, 2), dim3(32, 8), 0, stream>>>(x, xt);

  for (int blk = 0; blk < 2; ++blk) {
    const int base = 1 + blk * 12;
    const float* n1g = (const float*)d_in[base + 0];
    const float* n1b = (const float*)d_in[base + 1];
    const float* qw  = (const float*)d_in[base + 2];
    const float* rpb = (const float*)d_in[base + 3];
    const float* pw  = (const float*)d_in[base + 4];
    const float* pb  = (const float*)d_in[base + 5];
    const float* n2g = (const float*)d_in[base + 6];
    const float* n2b = (const float*)d_in[base + 7];
    const float* f1w = (const float*)d_in[base + 8];
    const float* f1b = (const float*)d_in[base + 9];
    const float* f2w = (const float*)d_in[base + 10];
    const float* f2b = (const float*)d_in[base + 11];

    cvt_f16<<<(384 * 128 + 255) / 256, 256, 0, stream>>>(qw,  wq, 384 * 128);
    cvt_f16<<<(128 * 128 + 255) / 256, 256, 0, stream>>>(pw,  wp, 128 * 128);
    cvt_f16<<<(512 * 128 + 255) / 256, 256, 0, stream>>>(f1w, w1, 512 * 128);
    cvt_f16<<<(128 * 512 + 255) / 256, 256, 0, stream>>>(f2w, w2, 128 * 512);

    // LN1 -> xn
    ln_kernel<<<T / 8, 256, 0, stream>>>(xt, n1g, n1b, xn);
    // qkv = xn @ qkv_w^T   (T x 384, f16)
    gemm_ws<384, 128, 0, false, false, true>
        <<<dim3(T / 256, 384 / 64), 256, 0, stream>>>(xn, wq, nullptr, nullptr,
                                                      big, nullptr);
    // windowed attention -> ao
    if (blk == 0)
      attn_kernel<0><<<dim3(NWIN, 4), 32, 0, stream>>>(big, rpb, ao);
    else
      attn_kernel<1><<<dim3(NWIN, 4), 32, 0, stream>>>(big, rpb, ao);
    // xr = xt + ao @ proj_w^T + proj_b   (f32)
    gemm_ws<128, 128, 0, true, true, false>
        <<<dim3(T / 256, 128 / 64), 256, 0, stream>>>(ao, wp, pb, xt,
                                                      nullptr, xr);
    // LN2 -> xn
    ln_kernel<<<T / 8, 256, 0, stream>>>(xr, n2g, n2b, xn);
    // h1 = gelu(xn @ fc1_w^T + fc1_b)   (T x 512, f16)
    gemm_ws<512, 128, 1, false, true, true>
        <<<dim3(T / 256, 512 / 64), 256, 0, stream>>>(xn, w1, f1b, nullptr,
                                                      big, nullptr);
    // xt = xr + h1 @ fc2_w^T + fc2_b    (f32)
    gemm_ws<128, 512, 0, true, true, false>
        <<<dim3(T / 256, 128 / 64), 256, 0, stream>>>(big, w2, f2b, xr,
                                                      nullptr, xt);
  }

  // (T,128) -> (B,C,D,H,W)
  from_tokens<<<dim3(NSPAT / 32, 4, 2), dim3(32, 8), 0, stream>>>(xt, (float*)d_out);
}
